// VectorQuantizer_44495861187031
// MI455X (gfx1250) — matrix operand order repaired and checked by reading, not compiled, not run on hardware
//
#include <hip/hip_runtime.h>

// ---------------------------------------------------------------------------
// VectorQuantizer forward for MI455X (gfx1250, wave32, WMMA, async-LDS DMA)
//   quantized[8,1024,1024], indices[8,1024], loss, logits[8,1024,8192]
// Distances via split-precision bf16 WMMA GEMM: z.e ~ zh.eh + zh.el + zl.eh
// Operands pre-split to bf16 hi/lo in workspace; GEMM stages tiles with
// GLOBAL_LOAD_ASYNC_TO_LDS_B128 double-buffered against the WMMA pipeline.
// ---------------------------------------------------------------------------

typedef __attribute__((ext_vector_type(16))) __bf16 v16bf;
typedef __attribute__((ext_vector_type(8)))  __bf16 v8bf;
typedef __attribute__((ext_vector_type(4)))  __bf16 v4bf;
typedef __attribute__((ext_vector_type(8)))  float  v8f;
typedef __attribute__((ext_vector_type(4)))  int    v4i;

#define N_TOK 8192   // B*S
#define DDIM  1024
#define KCB   8192
#define BM    128
#define BN    128
#define BK    32
#define LDSP  40     // padded LDS row stride in halves (80B: conflict-free, 16B aligned)

#if __has_builtin(__builtin_amdgcn_global_load_async_to_lds_b128)
#define HAS_ASYNC_LDS 1
typedef __attribute__((address_space(1))) v4i as1_v4i;   // global
typedef __attribute__((address_space(3))) v4i as3_v4i;   // LDS
#else
#define HAS_ASYNC_LDS 0
#endif

// --------------------------- helpers ---------------------------------------
__device__ inline void async_copy16(const __bf16* g, __bf16* l) {
#if HAS_ASYNC_LDS
    __builtin_amdgcn_global_load_async_to_lds_b128(
        (as1_v4i*)const_cast<__bf16*>(g), (as3_v4i*)l, 0, 0);
#else
    *(v8bf*)l = *(const v8bf*)g;   // synchronous fallback
#endif
}

__device__ inline void wait_async_le8() {
#if HAS_ASYNC_LDS
#if __has_builtin(__builtin_amdgcn_s_wait_asynccnt)
    __builtin_amdgcn_s_wait_asynccnt(8);
#else
    asm volatile("s_wait_asynccnt 0x8" ::: "memory");
#endif
#endif
}

__device__ inline void wait_async_le0() {
#if HAS_ASYNC_LDS
#if __has_builtin(__builtin_amdgcn_s_wait_asynccnt)
    __builtin_amdgcn_s_wait_asynccnt(0);
#else
    asm volatile("s_wait_asynccnt 0x0" ::: "memory");
#endif
#endif
}

__device__ inline unsigned int sortable_key(float d) {
    unsigned int u = __float_as_uint(d);
    return (u & 0x80000000u) ? ~u : (u | 0x80000000u);
}

__device__ inline unsigned long long shflxor64(unsigned long long x, int m) {
    unsigned int lo = (unsigned int)x;
    unsigned int hi = (unsigned int)(x >> 32);
    lo = __shfl_xor(lo, m, 32);
    hi = __shfl_xor(hi, m, 32);
    return ((unsigned long long)hi << 32) | lo;
}

// ---------------- prep: f32 -> (bf16 hi, bf16 lo) + row norm ----------------
__global__ __launch_bounds__(256)
void vq_prep(const float* __restrict__ src, __bf16* __restrict__ hi,
             __bf16* __restrict__ lo, float* __restrict__ norm) {
    const int row = blockIdx.x;
    const float4 v = ((const float4*)(src + (size_t)row * DDIM))[threadIdx.x];
    const float f[4] = {v.x, v.y, v.z, v.w};
    v4bf h4, l4;
    float s = 0.f;
#pragma unroll
    for (int j = 0; j < 4; ++j) {
        const __bf16 h = (__bf16)f[j];
        h4[j] = h;
        l4[j] = (__bf16)(f[j] - (float)h);
        s += f[j] * f[j];
    }
    *(v4bf*)(hi + (size_t)row * DDIM + threadIdx.x * 4) = h4;
    *(v4bf*)(lo + (size_t)row * DDIM + threadIdx.x * 4) = l4;
#pragma unroll
    for (int off = 16; off >= 1; off >>= 1) s += __shfl_xor(s, off, 32);
    __shared__ float ws[8];
    const int wid = threadIdx.x >> 5, lane = threadIdx.x & 31;
    if (lane == 0) ws[wid] = s;
    __syncthreads();
    if (threadIdx.x == 0) {
        float t = 0.f;
#pragma unroll
        for (int i = 0; i < 8; ++i) t += ws[i];
        norm[row] = t;
    }
}

// ------------------------------ main GEMM ----------------------------------
__global__ __launch_bounds__(256)
void vq_gemm_kernel(const __bf16* __restrict__ zhi, const __bf16* __restrict__ zlo,
                    const __bf16* __restrict__ ehi, const __bf16* __restrict__ elo,
                    const float* __restrict__ zn2, const float* __restrict__ en2,
                    unsigned long long* __restrict__ partial,
                    float* __restrict__ logits) {
    // double-buffered: [stage][array: zhi,zlo,ehi,elo][128 rows x 40 halves]
    __shared__ __bf16 sBuf[2][4][BM * LDSP];             // 80 KB
    __shared__ unsigned long long rowmin[BM];            // 1 KB

    const int tid     = threadIdx.x;
    const int colBlk  = blockIdx.x;           // tiles over KCB
    const int rowBlk  = blockIdx.y;           // tiles over N_TOK
    const int rowBase = rowBlk * BM;
    const int colBase = colBlk * BN;

    for (int i = tid; i < BM; i += 256) rowmin[i] = ~0ull;

    const int wave = tid >> 5, lane = tid & 31;
    const int wr = wave & 1, wc = wave >> 1;  // 2x4 wave grid: 64x32 tile per wave
    const int half = lane >> 4, l15 = lane & 15;

    const __bf16* gsrc[4] = {zhi, zlo, ehi, elo};
    const int gbase[4] = {rowBase, rowBase, colBase, colBase};

    // issue async stage for k0 = 0 into buffer 0  (8 x b128 per thread)
#pragma unroll
    for (int a = 0; a < 4; ++a) {
#pragma unroll
        for (int p = 0; p < 2; ++p) {
            const int c = tid + p * 256;          // 512 16B-chunks per array
            const int r = c >> 2, q = c & 3;
            async_copy16(gsrc[a] + (size_t)(gbase[a] + r) * DDIM + q * 8,
                         &sBuf[0][a][r * LDSP + q * 8]);
        }
    }

    v8f acc[4][2] = {};

    for (int k0 = 0; k0 < DDIM; k0 += BK) {
        const int s = (k0 / BK) & 1;
        if (k0 + BK < DDIM) {
            // issue next tile into the other buffer (overlaps with this tile's WMMAs)
#pragma unroll
            for (int a = 0; a < 4; ++a) {
#pragma unroll
                for (int p = 0; p < 2; ++p) {
                    const int c = tid + p * 256;
                    const int r = c >> 2, q = c & 3;
                    async_copy16(gsrc[a] + (size_t)(gbase[a] + r) * DDIM + (k0 + BK) + q * 8,
                                 &sBuf[s ^ 1][a][r * LDSP + q * 8]);
                }
            }
            wait_async_le8();   // current buffer's 8 copies (issued last iter) landed
        } else {
            wait_async_le0();
        }
        __syncthreads();        // all waves' portions of buffer s are in LDS

        // ---- load fragments (ISA 16-bit A / B wave32 layouts) ----
        // A 16x32: lanes 0-15 row=l, elems0-7=K0..7, elems8-15=K16..23; lanes16-31: K8..15/K24..31
        const __bf16* sZhi = &sBuf[s][0][0];
        const __bf16* sZlo = &sBuf[s][1][0];
        const __bf16* sEhi = &sBuf[s][2][0];
        const __bf16* sElo = &sBuf[s][3][0];

        v16bf ahi[4], alo[4];
#pragma unroll
        for (int mi = 0; mi < 4; ++mi) {
            const __bf16* bh = sZhi + (wr * 64 + mi * 16 + l15) * LDSP + half * 8;
            const v8bf h0 = *(const v8bf*)(bh);
            const v8bf h1 = *(const v8bf*)(bh + 16);
            ahi[mi] = __builtin_shufflevector(h0, h1, 0,1,2,3,4,5,6,7,8,9,10,11,12,13,14,15);
            const __bf16* bl = sZlo + (wr * 64 + mi * 16 + l15) * LDSP + half * 8;
            const v8bf l0 = *(const v8bf*)(bl);
            const v8bf l1 = *(const v8bf*)(bl + 16);
            alo[mi] = __builtin_shufflevector(l0, l1, 0,1,2,3,4,5,6,7,8,9,10,11,12,13,14,15);
        }
        // B 32x16: lanes 0-15 col=l, elems0-7=K0..7, elems8-15=K16..23; lanes16-31: K8..15/K24..31
        v16bf bhi[2], blo[2];
#pragma unroll
        for (int ni = 0; ni < 2; ++ni) {
            const __bf16* bh = sEhi + (wc * 32 + ni * 16 + l15) * LDSP + half * 8;
            const v8bf h0 = *(const v8bf*)(bh);
            const v8bf h1 = *(const v8bf*)(bh + 16);
            bhi[ni] = __builtin_shufflevector(h0, h1, 0,1,2,3,4,5,6,7,8,9,10,11,12,13,14,15);
            const __bf16* bl = sElo + (wc * 32 + ni * 16 + l15) * LDSP + half * 8;
            const v8bf l0 = *(const v8bf*)(bl);
            const v8bf l1 = *(const v8bf*)(bl + 16);
            blo[ni] = __builtin_shufflevector(l0, l1, 0,1,2,3,4,5,6,7,8,9,10,11,12,13,14,15);
        }

        // ---- split-precision accumulate: hi*hi + hi*lo + lo*hi ----
#pragma unroll
        for (int mi = 0; mi < 4; ++mi)
#pragma unroll
            for (int ni = 0; ni < 2; ++ni) {
                acc[mi][ni] = __builtin_amdgcn_wmma_f32_16x16x32_bf16(
                    false, ahi[mi], false, bhi[ni], (short)0, acc[mi][ni], false, false);
                acc[mi][ni] = __builtin_amdgcn_wmma_f32_16x16x32_bf16(
                    false, ahi[mi], false, blo[ni], (short)0, acc[mi][ni], false, false);
                acc[mi][ni] = __builtin_amdgcn_wmma_f32_16x16x32_bf16(
                    false, alo[mi], false, bhi[ni], (short)0, acc[mi][ni], false, false);
            }
        __syncthreads();        // all waves done reading buffer s before it is re-issued
    }

    // ---- epilogue: distances, logits, per-row argmin key reduction ----
    float en2v[2];
#pragma unroll
    for (int ni = 0; ni < 2; ++ni) en2v[ni] = en2[colBase + wc * 32 + ni * 16 + l15];

#pragma unroll
    for (int mi = 0; mi < 4; ++mi) {
#pragma unroll
        for (int v = 0; v < 8; ++v) {
            const int mloc = wr * 64 + mi * 16 + half * 8 + v;  // C: lanes16-31 -> M = v+8
            const int gm = rowBase + mloc;
            const float zz = zn2[gm];
            unsigned long long best = ~0ull;
#pragma unroll
            for (int ni = 0; ni < 2; ++ni) {
                const int gn = colBase + wc * 32 + ni * 16 + l15;
                const float dot = acc[mi][ni][v];
                const float dist = zz + en2v[ni] - 2.0f * dot;
                logits[(size_t)gm * KCB + gn] = -dist;
                const unsigned long long key =
                    ((unsigned long long)sortable_key(dist) << 32) | (unsigned int)gn;
                best = key < best ? key : best;
            }
#pragma unroll
            for (int m = 1; m <= 8; m <<= 1) {
                const unsigned long long o = shflxor64(best, m);
                best = o < best ? o : best;
            }
            if (l15 == 0) atomicMin(&rowmin[mloc], best);
        }
    }
    __syncthreads();
    for (int i = tid; i < BM; i += 256)
        partial[(size_t)colBlk * N_TOK + rowBase + i] = rowmin[i];
}

// ------------------------------ finalize -----------------------------------
__global__ __launch_bounds__(256)
void vq_finalize_kernel(const float* __restrict__ cb,
                        const unsigned long long* __restrict__ partial,
                        float* __restrict__ quant, float* __restrict__ indices,
                        float* __restrict__ loss) {
    const int n = blockIdx.x;
    __shared__ int sidx;
    if (threadIdx.x == 0) {
        unsigned long long best = ~0ull;
        for (int c = 0; c < KCB / BN; ++c) {
            const unsigned long long k = partial[(size_t)c * N_TOK + n];
            best = k < best ? k : best;
        }
        const int idx = (int)(best & 0xFFFFFFFFull);
        sidx = idx;
        indices[n] = (float)idx;
        if (n == 0) loss[0] = 0.0f;
    }
    __syncthreads();
    const int idx = sidx;
    const float4* src = (const float4*)(cb + (size_t)idx * DDIM);
    float4* dst = (float4*)(quant + (size_t)n * DDIM);
    dst[threadIdx.x] = src[threadIdx.x];  // 256 * 4 floats = full row
}

// ------------------------------ launcher -----------------------------------
extern "C" void kernel_launch(void* const* d_in, const int* in_sizes, int n_in,
                              void* d_out, int out_size, void* d_ws, size_t ws_size,
                              hipStream_t stream) {
    (void)in_sizes; (void)n_in; (void)out_size; (void)ws_size;
    const float* z  = (const float*)d_in[0];   // [8,1024,1024] f32
    const float* cb = (const float*)d_in[1];   // [8192,1024]  f32

    float* out     = (float*)d_out;
    float* quant   = out;                       // 8*1024*1024
    float* indices = out + 8388608;             // 8192
    float* loss    = out + 8396800;             // 1
    float* logits  = out + 8396801;             // 8192*8192

    // workspace: bf16 hi/lo operands (64MB) + norms + argmin partials
    __bf16* zhi = (__bf16*)d_ws;                          // 8192*1024
    __bf16* zlo = zhi + (size_t)N_TOK * DDIM;
    __bf16* ehi = zlo + (size_t)N_TOK * DDIM;
    __bf16* elo = ehi + (size_t)KCB * DDIM;
    float*  zn2 = (float*)(elo + (size_t)KCB * DDIM);     // @64MB
    float*  en2 = zn2 + N_TOK;
    unsigned long long* partial =
        (unsigned long long*)((char*)d_ws + (64u << 20) + 65536);  // [64][8192] u64

    vq_prep<<<N_TOK, 256, 0, stream>>>(z, zhi, zlo, zn2);
    vq_prep<<<KCB, 256, 0, stream>>>(cb, ehi, elo, en2);
    vq_gemm_kernel<<<dim3(KCB / BN, N_TOK / BM), 256, 0, stream>>>(
        zhi, zlo, ehi, elo, zn2, en2, partial, logits);
    vq_finalize_kernel<<<N_TOK, 256, 0, stream>>>(cb, partial, quant, indices, loss);
}
